// SuperPointMatching_30657476558961
// MI455X (gfx1250) — matrix operand order repaired and checked by reading, not compiled, not run on hardware
//
#include <hip/hip_runtime.h>
#include <hip/hip_bf16.h>

// ---------------------------------------------------------------------------
// SuperPoint matching on MI455X (gfx1250, wave32, WMMA)
//   scores = exp(2*ref@src^T - 2)  (GEMM via v_wmma_f32_16x16x32_f16)
//   dual normalization via row/col sums (fused atomics in GEMM epilogue)
//   global top-256 via float-bit radix select + rank sort
// ---------------------------------------------------------------------------

typedef __attribute__((ext_vector_type(16))) _Float16 v16h;
typedef __attribute__((ext_vector_type(8)))  float    v8f;

#define NN 4096
#define MM 4096
#define DD 256
#define TOPK 256
#define NBUCKETS 65536

// ---------------- init: zero accumulators, poison candidates ----------------
__global__ __launch_bounds__(256)
void spm_init_kernel(float* row_sum, float* col_sum, unsigned* hist,
                     unsigned* cnt, float* cand_val, int* cand_idx) {
    int i = blockIdx.x * blockDim.x + threadIdx.x;
    if (i < NN) row_sum[i] = 0.0f;
    if (i < MM) col_sum[i] = 0.0f;
    if (i < NBUCKETS) hist[i] = 0u;
    if (i < 4) cnt[i] = 0u;
    if (i < TOPK) { cand_val[i] = -1.0f; cand_idx[i] = 0; }
}

// ---------------- f32 -> f16 feature conversion -----------------------------
__global__ __launch_bounds__(256)
void spm_convert_kernel(const float* __restrict__ ref, const float* __restrict__ src,
                        _Float16* __restrict__ refh, _Float16* __restrict__ srch) {
    const int total = NN * DD;
    for (int i = blockIdx.x * blockDim.x + threadIdx.x; i < total;
         i += gridDim.x * blockDim.x) {
        refh[i] = (_Float16)ref[i];
        srch[i] = (_Float16)src[i];
    }
}

// ---------------- GEMM + exp epilogue + row/col sums ------------------------
// One wave per 16x16 output tile. blockDim = 256 (8 waves).
__global__ __launch_bounds__(256)
void spm_gemm_exp_kernel(const _Float16* __restrict__ refh,
                         const _Float16* __restrict__ srch,
                         const unsigned char* __restrict__ rmask,
                         const unsigned char* __restrict__ smask,
                         float* __restrict__ scores,
                         float* __restrict__ row_sum,
                         float* __restrict__ col_sum) {
    const int lane = threadIdx.x & 31;
    const int wid  = blockIdx.x * 8 + (threadIdx.x >> 5);
    const int tiles_n = MM / 16;              // 256
    const int tile_m = wid / tiles_n;
    const int tile_n = wid % tiles_n;

    const int r  = lane & 15;                 // row-in-tile (A) / col-in-tile (B)
    const int hi = lane >> 4;                 // half-wave select

    const _Float16* aRow = refh + (size_t)(tile_m * 16 + r) * DD;
    const _Float16* bRow = srch + (size_t)(tile_n * 16 + r) * DD;

    v8f acc = {};
    #pragma unroll
    for (int kk = 0; kk < DD / 32; ++kk) {
        union { v16h v; unsigned u[8]; } a, b;

        // A fragment (16x32 f16): lane<16 -> K in {kk*32+0..7, +16..23}
        //                         lane>=16 -> K in {kk*32+8..15, +24..31}
        const int ka = kk * 32 + hi * 8;
        const unsigned* ap  = reinterpret_cast<const unsigned*>(aRow + ka);
        const unsigned* ap2 = reinterpret_cast<const unsigned*>(aRow + ka + 16);
        #pragma unroll
        for (int j = 0; j < 4; ++j) { a.u[j] = ap[j]; a.u[4 + j] = ap2[j]; }

        // B fragment (32x16 f16): lane<16 -> K in {kk*32+0..7, +16..23}
        //                         lane>=16 -> K in {kk*32+8..15, +24..31}
        const int kb = kk * 32 + hi * 8;
        const unsigned* bp = reinterpret_cast<const unsigned*>(bRow + kb);
        #pragma unroll
        for (int j = 0; j < 8; ++j) b.u[j] = bp[(j & 3) + (j >> 2) * 8];

        if (kk + 1 < DD / 32) {
            __builtin_prefetch(aRow + (kk + 1) * 32, 0, 1);
            __builtin_prefetch(bRow + (kk + 1) * 32, 0, 1);
        }

        acc = __builtin_amdgcn_wmma_f32_16x16x32_f16(
            /*neg_a=*/false, a.v, /*neg_b=*/false, b.v,
            /*c_mod=*/(short)0, acc, /*reuse_a=*/false, /*reuse_b=*/false);
    }

    // Epilogue: C/D layout -> lane group 0-15 holds M=v, group 16-31 holds M=8+v,
    // column N = lane%16.
    const int col = tile_n * 16 + r;
    const bool cm = smask[col] != 0;
    float colPart = 0.0f;
    #pragma unroll
    for (int v = 0; v < 8; ++v) {
        const int row = tile_m * 16 + v + 8 * hi;
        const bool rm = rmask[row] != 0;
        float s = (rm && cm) ? __expf(2.0f * acc[v] - 2.0f) : 0.0f;
        scores[(size_t)row * MM + col] = s;
        colPart += s;
        // reduce across the 16 lanes sharing this row
        float rs = s;
        rs += __shfl_xor(rs, 8, 16);
        rs += __shfl_xor(rs, 4, 16);
        rs += __shfl_xor(rs, 2, 16);
        rs += __shfl_xor(rs, 1, 16);
        if (r == 0) atomicAdd(&row_sum[row], rs);
    }
    // combine the two half-waves that hold the same column
    colPart += __shfl_xor(colPart, 16, 32);
    if (hi == 0) atomicAdd(&col_sum[col], colPart);
}

// ---------------- normalization value + histogram ---------------------------
__device__ __forceinline__ float spm_norm_val(float s, float rs, float cs) {
    return (s * s) / (rs * cs);
}

__global__ __launch_bounds__(256)
void spm_hist_kernel(const float* __restrict__ scores,
                     const float* __restrict__ row_sum,
                     const float* __restrict__ col_sum,
                     unsigned* __restrict__ hist) {
    const unsigned total = (unsigned)NN * (unsigned)MM;
    for (unsigned i = blockIdx.x * blockDim.x + threadIdx.x; i < total;
         i += gridDim.x * blockDim.x) {
        float s = scores[i];
        if (s <= 0.0f) continue;                  // masked-out pair
        unsigned rr = i / MM, cc = i % MM;
        float v = spm_norm_val(s, row_sum[rr], col_sum[cc]);
        unsigned b = __float_as_uint(v) >> 16;    // monotone for v >= 0
        atomicAdd(&hist[b], 1u);
    }
}

// ---------------- find threshold bucket (single block) ----------------------
__global__ __launch_bounds__(256)
void spm_select_kernel(const unsigned* __restrict__ hist, unsigned* __restrict__ params) {
    __shared__ unsigned chunk[256];
    const int t = threadIdx.x;
    unsigned s = 0;
    const int base = t * (NBUCKETS / 256);
    for (int j = 0; j < NBUCKETS / 256; ++j) s += hist[base + j];
    chunk[t] = s;
    __syncthreads();
    if (t == 0) {
        unsigned cum = 0;
        unsigned T = 0, above = 0;
        bool found = false;
        for (int c = 255; c >= 0 && !found; --c) {
            if (cum + chunk[c] >= TOPK) {
                unsigned cum2 = cum;
                for (int b = c * 256 + 255; b >= c * 256; --b) {
                    unsigned h = hist[b];
                    if (cum2 + h >= TOPK) { T = (unsigned)b; above = cum2; found = true; break; }
                    cum2 += h;
                }
                if (!found) { T = (unsigned)(c * 256); above = cum2; found = true; }
            } else {
                cum += chunk[c];
            }
        }
        if (!found) { T = 0; above = cum; }       // fewer than TOPK valid entries
        params[0] = T;
        params[1] = above;
    }
}

// ---------------- collect top-k candidates ----------------------------------
__global__ __launch_bounds__(256)
void spm_collect_kernel(const float* __restrict__ scores,
                        const float* __restrict__ row_sum,
                        const float* __restrict__ col_sum,
                        const unsigned* __restrict__ params,
                        unsigned* __restrict__ cnt,
                        float* __restrict__ cand_val,
                        int* __restrict__ cand_idx) {
    const unsigned T = params[0];
    const unsigned above = params[1];
    const unsigned total = (unsigned)NN * (unsigned)MM;
    for (unsigned i = blockIdx.x * blockDim.x + threadIdx.x; i < total;
         i += gridDim.x * blockDim.x) {
        float s = scores[i];
        if (s <= 0.0f) continue;
        unsigned rr = i / MM, cc = i % MM;
        float v = spm_norm_val(s, row_sum[rr], col_sum[cc]);
        unsigned b = __float_as_uint(v) >> 16;
        if (b > T) {
            unsigned p = atomicAdd(&cnt[0], 1u);
            if (p < TOPK) { cand_val[p] = v; cand_idx[p] = (int)i; }
        } else if (b == T) {
            unsigned p = atomicAdd(&cnt[1], 1u);
            unsigned slot = above + p;
            if (slot < TOPK) { cand_val[slot] = v; cand_idx[slot] = (int)i; }
        }
    }
}

// ---------------- rank-sort candidates, emit outputs ------------------------
// d_out layout (reference return order): int32 ref_idx[256], int32 src_idx[256],
// float corr_scores[256].
__global__ __launch_bounds__(256)
void spm_emit_kernel(const float* __restrict__ cand_val,
                     const int* __restrict__ cand_idx,
                     float* __restrict__ out) {
    __shared__ float v[TOPK];
    __shared__ int   id[TOPK];
    const int t = threadIdx.x;
    v[t] = cand_val[t];
    id[t] = cand_idx[t];
    __syncthreads();
    const float mv = v[t];
    const int   mi = id[t];
    int rank = 0;
    for (int j = 0; j < TOPK; ++j) {
        if (v[j] > mv || (v[j] == mv && id[j] < mi)) ++rank;
    }
    int* oi = (int*)out;
    oi[rank]         = mi / MM;   // ref index
    oi[TOPK + rank]  = mi % MM;   // src index
    out[2 * TOPK + rank] = mv;    // score
}

// ---------------------------------------------------------------------------
extern "C" void kernel_launch(void* const* d_in, const int* in_sizes, int n_in,
                              void* d_out, int out_size, void* d_ws, size_t ws_size,
                              hipStream_t stream) {
    const float* ref = (const float*)d_in[0];
    const float* src = (const float*)d_in[1];
    const unsigned char* rmask = (const unsigned char*)d_in[2];
    const unsigned char* smask = (const unsigned char*)d_in[3];
    float* out = (float*)d_out;

    char* ws = (char*)d_ws;
    size_t off = 0;
    float*    scores  = (float*)(ws + off);    off += (size_t)NN * MM * sizeof(float);   // 64 MB
    _Float16* refh    = (_Float16*)(ws + off); off += (size_t)NN * DD * sizeof(_Float16);
    _Float16* srch    = (_Float16*)(ws + off); off += (size_t)MM * DD * sizeof(_Float16);
    float*    row_sum = (float*)(ws + off);    off += NN * sizeof(float);
    float*    col_sum = (float*)(ws + off);    off += MM * sizeof(float);
    unsigned* hist    = (unsigned*)(ws + off); off += NBUCKETS * sizeof(unsigned);
    unsigned* cnt     = (unsigned*)(ws + off); off += 4 * sizeof(unsigned);
    unsigned* params  = (unsigned*)(ws + off); off += 4 * sizeof(unsigned);
    float*    cand_v  = (float*)(ws + off);    off += TOPK * sizeof(float);
    int*      cand_i  = (int*)(ws + off);      off += TOPK * sizeof(int);

    // 1) init accumulators
    spm_init_kernel<<<NBUCKETS / 256, 256, 0, stream>>>(row_sum, col_sum, hist,
                                                        cnt, cand_v, cand_i);
    // 2) f32 -> f16 features
    spm_convert_kernel<<<2048, 256, 0, stream>>>(ref, src, refh, srch);
    // 3) WMMA GEMM + exp + row/col sums; 65536 tiles, 8 waves/block
    spm_gemm_exp_kernel<<<(NN / 16) * (MM / 16) / 8, 256, 0, stream>>>(
        refh, srch, rmask, smask, scores, row_sum, col_sum);
    // 4) histogram of normalized scores
    spm_hist_kernel<<<4096, 256, 0, stream>>>(scores, row_sum, col_sum, hist);
    // 5) threshold bucket
    spm_select_kernel<<<1, 256, 0, stream>>>(hist, params);
    // 6) gather candidates
    spm_collect_kernel<<<4096, 256, 0, stream>>>(scores, row_sum, col_sum,
                                                 params, cnt, cand_v, cand_i);
    // 7) rank-sort + emit
    spm_emit_kernel<<<1, 256, 0, stream>>>(cand_v, cand_i, out);
}
